// MSA_Stack_8538394985034
// MI455X (gfx1250) — matrix operand order repaired and checked by reading, not compiled, not run on hardware
//
#include <hip/hip_runtime.h>

typedef __bf16 bf16;
typedef __bf16 v8bf  __attribute__((ext_vector_type(8)));
typedef __bf16 v16bf __attribute__((ext_vector_type(16)));
typedef float  v8f   __attribute__((ext_vector_type(8)));
typedef int    v4i   __attribute__((ext_vector_type(4)));

#define LN_EPS 1e-5f

// ---------------------------------------------------------------- helpers
__device__ __forceinline__ v16bf cat8(v8bf lo, v8bf hi) {
  return __builtin_shufflevector(lo, hi, 0,1,2,3,4,5,6,7,8,9,10,11,12,13,14,15);
}
__device__ __forceinline__ v8f wmma_bf16(v16bf a, v16bf b, v8f c) {
  // v_wmma_f32_16x16x32_bf16: D = A(16x32) * B(32x16) + C
  return __builtin_amdgcn_wmma_f32_16x16x32_bf16(false, a, false, b, (short)0, c,
                                                 false, false);
}
__device__ __forceinline__ float rmax16(float v) {
#pragma unroll
  for (int m = 1; m < 16; m <<= 1) v = fmaxf(v, __shfl_xor(v, m, 32));
  return v;
}
__device__ __forceinline__ float rsum16(float v) {
#pragma unroll
  for (int m = 1; m < 16; m <<= 1) v += __shfl_xor(v, m, 32);
  return v;
}
__device__ __forceinline__ float rsum32(float v) {
#pragma unroll
  for (int m = 1; m < 32; m <<= 1) v += __shfl_xor(v, m, 32);
  return v;
}

// ------------------------------------------------- weight convert+transpose
// W[K][N] (f32, row major) -> WT[N][K] (bf16, row major)
__global__ void wconv_k(const float* __restrict__ W, bf16* __restrict__ WT,
                        int K, int N) {
  int i = blockIdx.x * 256 + threadIdx.x;
  if (i >= K * N) return;
  int n = i / K, k = i % K;
  WT[i] = (bf16)W[(long)k * N + n];
}

// ------------------------------------------------------------- LayerNorm
// rows of 256, one wave per row, bf16 output
__global__ __launch_bounds__(256)
void ln_k(const float* __restrict__ in, bf16* __restrict__ out,
          const float* __restrict__ gw, const float* __restrict__ bw) {
  int wid = threadIdx.x >> 5, lane = threadIdx.x & 31;
  long row = (long)blockIdx.x * 8 + wid;
  const float* p = in + row * 256;
  float x[8], s = 0.f, s2 = 0.f;
#pragma unroll
  for (int i = 0; i < 8; ++i) { x[i] = p[lane + i*32]; s += x[i]; s2 += x[i]*x[i]; }
  s  = rsum32(s);
  s2 = rsum32(s2);
  float mean = s * (1.f/256.f);
  float var  = s2 * (1.f/256.f) - mean * mean;
  float inv  = rsqrtf(var + LN_EPS);
  bf16* o = out + row * 256;
#pragma unroll
  for (int i = 0; i < 8; ++i) {
    int c = lane + i*32;
    o[c] = (bf16)((x[i]-mean)*inv*gw[c] + bw[c]);
  }
}

// ------------------------------------------- pair bias: LN(z) @ Wb + bb
// z: [256*256][128] f32 ; pb: [8][256][256] f32
__global__ __launch_bounds__(256)
void pbias_k(const float* __restrict__ z, const float* __restrict__ gw,
             const float* __restrict__ bw, const float* __restrict__ Wb,
             const float* __restrict__ bb, float* __restrict__ pb) {
  int wid = threadIdx.x >> 5, lane = threadIdx.x & 31;
  long pos = (long)blockIdx.x * 8 + wid;          // = i*256 + j
  const float* zp = z + pos * 128;
  float zn[4], s = 0.f, s2 = 0.f;
#pragma unroll
  for (int k = 0; k < 4; ++k) { zn[k] = zp[lane + k*32]; s += zn[k]; s2 += zn[k]*zn[k]; }
  s  = rsum32(s);
  s2 = rsum32(s2);
  float mean = s * (1.f/128.f), var = s2 * (1.f/128.f) - mean*mean;
  float inv = rsqrtf(var + LN_EPS);
#pragma unroll
  for (int k = 0; k < 4; ++k) { int c = lane + k*32; zn[k] = (zn[k]-mean)*inv*gw[c] + bw[c]; }
  float acc[8];
#pragma unroll
  for (int h = 0; h < 8; ++h) {
    float a = 0.f;
#pragma unroll
    for (int k = 0; k < 4; ++k) a += zn[k] * Wb[(lane + k*32)*8 + h];
    acc[h] = rsum32(a);
  }
  if (lane == 0) {
#pragma unroll
    for (int h = 0; h < 8; ++h) pb[(long)h*65536 + pos] = acc[h] + bb[h];
  }
}

// ---------------------------------------------------------- WMMA GEMM
// C[M,N] = A[M,K](bf16) * BT[N,K]^T(bf16) ; 128x128 block tile, 8 waves.
enum { EPI_QKV = 0, EPI_PROJ1 = 1, EPI_PROJ2 = 2, EPI_MLP1 = 3, EPI_MLP2 = 4 };

template<int EPI>
__global__ __launch_bounds__(256)
void gemm_k(const bf16* __restrict__ A, const bf16* __restrict__ BT,
            int N, int K, const float* __restrict__ bias,
            const bf16* __restrict__ addbf, const float* __restrict__ addf,
            float* __restrict__ outf, bf16* outbf) {
  int wid  = threadIdx.x >> 5, lane = threadIdx.x & 31;
  int half = lane >> 4, l16 = lane & 15;
  int m_base = blockIdx.y * 128 + (wid & 3) * 32;
  int n_base = blockIdx.x * 128 + (wid >> 2) * 64;

  v8f acc[2][4] = {};
  const bf16* a0 = A  + (long)(m_base +      l16) * K + 8  * half;
  const bf16* a1 = A  + (long)(m_base + 16 + l16) * K + 8  * half;
  const bf16* bp = BT + (long)(n_base +      l16) * K + 8  * half;

  for (int k0 = 0; k0 < K; k0 += 32) {
    if (k0 + 32 < K) {                 // global_prefetch_b8 hints
      __builtin_prefetch(a0 + k0 + 32, 0, 3);
      __builtin_prefetch(bp + k0 + 32, 0, 3);
    }
    v16bf af[2], bfr[4];
    af[0] = cat8(*(const v8bf*)(a0 + k0), *(const v8bf*)(a0 + k0 + 16));
    af[1] = cat8(*(const v8bf*)(a1 + k0), *(const v8bf*)(a1 + k0 + 16));
#pragma unroll
    for (int nt = 0; nt < 4; ++nt)
      bfr[nt] = cat8(*(const v8bf*)(bp + (long)nt * 16 * K + k0), *(const v8bf*)(bp + (long)nt * 16 * K + k0 + 16));
#pragma unroll
    for (int mt = 0; mt < 2; ++mt)
#pragma unroll
      for (int nt = 0; nt < 4; ++nt)
        acc[mt][nt] = wmma_bf16(af[mt], bfr[nt], acc[mt][nt]);
  }

#pragma unroll
  for (int mt = 0; mt < 2; ++mt) {
#pragma unroll
    for (int nt = 0; nt < 4; ++nt) {
#pragma unroll
      for (int r = 0; r < 8; ++r) {
        int row = m_base + mt*16 + r + 8*half;
        int col = n_base + nt*16 + l16;
        long idx = (long)row * N + col;
        float v = acc[mt][nt][r];
        if (EPI == EPI_QKV) {
          // qkv col c = d*32 + k4*8 + h  ->  store as [row][k4][h][d]
          int h = col & 7, k4 = (col >> 3) & 3, d = col >> 5;
          outbf[(long)row * 1024 + k4*256 + h*32 + d] = (bf16)v;
        } else if (EPI == EPI_PROJ1) {
          v += (float)addbf[idx];           // residual adds ln1(x)
          outf[idx] = v;
        } else if (EPI == EPI_PROJ2) {
          v += (float)outbf[idx];           // residual adds ln3(x); same buf
          outf[idx] = v;                    // X (f32) for final residual
          outbf[idx] = (bf16)v;             // bf16 copy feeds the MLP
        } else if (EPI == EPI_MLP1) {
          v += bias[col];
          v = fmaxf(v, 0.f);
          outbf[idx] = (bf16)v;
        } else {                            // EPI_MLP2
          v += bias[col] + addf[idx];
          outf[idx] = v;
        }
      }
    }
  }
}

// --------------------------------------------- flash-attention with WMMA
// qkv layout: [row][k4][h][32] bf16 (k4: 0=q 1=k 2=v 3=g)
// one block per (n, h, rowchunk); 8 waves x 16 rows; square T x T attention
template<int USE_BIAS>
__global__ __launch_bounds__(256)
void attn_k(const bf16* __restrict__ qkv, const float* __restrict__ pb,
            bf16* __restrict__ ao, int T, int nblk, long row0_mult,
            long row_stride) {
  __shared__ __attribute__((aligned(64))) bf16 vT[32][256];     // V transposed
  __shared__ __attribute__((aligned(64))) bf16 plds[8][16][32]; // probs bounce
#if __has_builtin(__builtin_amdgcn_global_load_async_to_lds_b128)
  __shared__ __attribute__((aligned(64))) bf16 vrow[256][32];   // V row-major
#endif
  int tid = threadIdx.x;
  int wid = tid >> 5, lane = tid & 31, half = lane >> 4, l16 = lane & 15;
  int bid = blockIdx.x;
  int blk = bid % nblk;
  int h   = (bid / nblk) & 7;
  int n   = bid / (nblk * 8);
  long row0 = (long)n * row0_mult;
  const long LQ = 1024;

#if __has_builtin(__builtin_amdgcn_global_load_async_to_lds_b128)
  // Stage V rows via ASYNCcnt DMA (no VGPR round trip), then transpose in LDS.
  for (int base = tid * 8; base < T * 32; base += 256 * 8) {
    int j = base >> 5, d0 = base & 31;
    const bf16* g = qkv + (row0 + (long)j * row_stride) * LQ + 512 + h*32 + d0;
    __builtin_amdgcn_global_load_async_to_lds_b128(
        (v4i*)(void*)g, (v4i*)(void*)&vrow[j][d0], 0, 0);
  }
  // per-wave async completion BEFORE the workgroup barrier
  asm volatile("s_wait_asynccnt 0" ::: "memory");
  __syncthreads();
  for (int idx = tid; idx < T * 32; idx += 256) {
    int d = idx & 31, j = idx >> 5;
    vT[d][j] = vrow[j][d];
  }
#else
  // fallback: vectorized global loads + LDS transpose scatter
  for (int base = tid * 8; base < T * 32; base += 256 * 8) {
    int j = base >> 5, d0 = base & 31;
    v8bf vv = *(const v8bf*)(qkv + (row0 + (long)j * row_stride) * LQ + 512 + h*32 + d0);
#pragma unroll
    for (int e = 0; e < 8; ++e) vT[d0 + e][j] = vv[e];
  }
#endif
  __syncthreads();

  int t0 = blk * 128 + wid * 16;
  const bf16* qrow = qkv + (row0 + (long)(t0 + l16) * row_stride) * LQ + h*32;
  v16bf qf = cat8(*(const v8bf*)(qrow + 16*half),
                  *(const v8bf*)(qrow + 16*half + 8));

  float mrun[8], lrun[8];
#pragma unroll
  for (int r = 0; r < 8; ++r) { mrun[r] = -1e30f; lrun[r] = 0.f; }
  v8f o0 = {}, o1 = {};
  const float scale = 0.17677669529663687f;   // 32^-0.5

  for (int jb = 0; jb < T; jb += 32) {
    // S tiles: q(16x32) * k^T ; k rows are d-contiguous => direct B frags
    const bf16* k0p = qkv + (row0 + (long)(jb      + l16) * row_stride) * LQ + 256 + h*32 + 16*half;
    const bf16* k1p = qkv + (row0 + (long)(jb + 16 + l16) * row_stride) * LQ + 256 + h*32 + 16*half;
    v16bf kf0 = *(const v16bf*)k0p;
    v16bf kf1 = *(const v16bf*)k1p;
    v8f s0 = {}, s1 = {};
    s0 = wmma_bf16(qf, kf0, s0);
    s1 = wmma_bf16(qf, kf1, s1);

#pragma unroll
    for (int r = 0; r < 8; ++r) {
      float v0 = s0[r] * scale, v1 = s1[r] * scale;
      if (USE_BIAS) {
        int ti = t0 + r + 8*half;
        v0 += pb[((long)h * T + ti) * T + jb + l16];
        v1 += pb[((long)h * T + ti) * T + jb + 16 + l16];
      }
      float bm = rmax16(fmaxf(v0, v1));
      float mn = fmaxf(mrun[r], bm);
      float al = __expf(mrun[r] - mn);
      float p0 = __expf(v0 - mn), p1 = __expf(v1 - mn);
      float rs = rsum16(p0 + p1);
      lrun[r] = lrun[r] * al + rs;
      mrun[r] = mn;
      o0[r] *= al;
      o1[r] *= al;
      plds[wid][r + 8*half][l16]      = (bf16)p0;
      plds[wid][r + 8*half][16 + l16] = (bf16)p1;
    }
    asm volatile("s_wait_dscnt 0" ::: "memory");

    // reshape probs C-layout -> A fragment, then O += P(16x32) @ V(32x32)
    v16bf pf  = cat8(*(const v8bf*)&plds[wid][l16][16*half],
                     *(const v8bf*)&plds[wid][l16][16*half + 8]);
    v16bf vf0 = *(const v16bf*)&vT[l16][jb + 16*half];
    v16bf vf1 = *(const v16bf*)&vT[16 + l16][jb + 16*half];
    o0 = wmma_bf16(pf, vf0, o0);
    o1 = wmma_bf16(pf, vf1, o1);
  }

  // normalize, sigmoid gate, scatter to [row][h*32+d]
#pragma unroll
  for (int r = 0; r < 8; ++r) {
    float inv = 1.0f / lrun[r];
    int ti = t0 + r + 8*half;
    long grow = row0 + (long)ti * row_stride;
    const bf16* gp = qkv + grow * LQ + 768 + h*32;
    float g0 = (float)gp[l16],      g1 = (float)gp[16 + l16];
    g0 = 1.0f / (1.0f + __expf(-g0));
    g1 = 1.0f / (1.0f + __expf(-g1));
    bf16* op = ao + grow * 256 + h*32;
    op[l16]      = (bf16)(o0[r] * inv * g0);
    op[16 + l16] = (bf16)(o1[r] * inv * g1);
  }
}

// ------------------------------------------------------------------ host
extern "C" void kernel_launch(void* const* d_in, const int* in_sizes, int n_in,
                              void* d_out, int out_size, void* d_ws, size_t ws_size,
                              hipStream_t stream) {
  (void)in_sizes; (void)n_in; (void)out_size; (void)ws_size;
  const float* x_in   = (const float*)d_in[0];
  const float* z_in   = (const float*)d_in[1];
  const float* Wqkv_r = (const float*)d_in[2];
  const float* W0_r   = (const float*)d_in[3];
  const float* Wb     = (const float*)d_in[4];
  const float* bb     = (const float*)d_in[5];
  const float* Wqkv_c = (const float*)d_in[6];
  const float* W0_c   = (const float*)d_in[7];
  const float* W1     = (const float*)d_in[8];
  const float* b1     = (const float*)d_in[9];
  const float* W2     = (const float*)d_in[10];
  const float* b2     = (const float*)d_in[11];
  const float* g1     = (const float*)d_in[12];
  const float* be1    = (const float*)d_in[13];
  const float* g2     = (const float*)d_in[14];
  const float* be2    = (const float*)d_in[15];
  const float* g3     = (const float*)d_in[16];
  const float* be3    = (const float*)d_in[17];
  float* out = (float*)d_out;

  const int M = 32768;                 // 128 seqs * 256 residues
  char* ws = (char*)d_ws;
  size_t off = 0;
  auto alloc = [&](size_t bytes) {
    char* p = ws + off;
    off = (off + bytes + 255) & ~(size_t)255;
    return p;
  };
  bf16*  XN  = (bf16*) alloc((size_t)M * 256 * 2);   // LN output (bf16)
  bf16*  QKV = (bf16*) alloc((size_t)M * 1024 * 2);  // qkv / MLP hidden
  bf16*  AO  = (bf16*) alloc((size_t)M * 256 * 2);   // gated attention out
  float* X   = (float*)alloc((size_t)M * 256 * 4);   // fp32 residual stream
  float* PB  = (float*)alloc((size_t)8 * 256 * 256 * 4);
  bf16*  TQr = (bf16*) alloc((size_t)1024 * 256 * 2);
  bf16*  T0r = (bf16*) alloc((size_t)256 * 256 * 2);
  bf16*  TQc = (bf16*) alloc((size_t)1024 * 256 * 2);
  bf16*  T0c = (bf16*) alloc((size_t)256 * 256 * 2);
  bf16*  T1  = (bf16*) alloc((size_t)1024 * 256 * 2);
  bf16*  T2  = (bf16*) alloc((size_t)256 * 1024 * 2);

  dim3 blk(256);
  // weights -> bf16, transposed [N][K]
  wconv_k<<<(256*1024+255)/256, blk, 0, stream>>>(Wqkv_r, TQr, 256, 1024);
  wconv_k<<<(256*256 +255)/256, blk, 0, stream>>>(W0_r,   T0r, 256, 256);
  wconv_k<<<(256*1024+255)/256, blk, 0, stream>>>(Wqkv_c, TQc, 256, 1024);
  wconv_k<<<(256*256 +255)/256, blk, 0, stream>>>(W0_c,   T0c, 256, 256);
  wconv_k<<<(256*1024+255)/256, blk, 0, stream>>>(W1,     T1,  256, 1024);
  wconv_k<<<(1024*256+255)/256, blk, 0, stream>>>(W2,     T2, 1024, 256);

  // ln1 + pair bias
  ln_k<<<M/8, blk, 0, stream>>>(x_in, XN, g1, be1);
  pbias_k<<<65536/8, blk, 0, stream>>>(z_in, g2, be2, Wb, bb, PB);

  // row-wise gated MHSA with pair bias
  gemm_k<EPI_QKV><<<dim3(8, M/128), blk, 0, stream>>>(
      XN, TQr, 1024, 256, nullptr, nullptr, nullptr, nullptr, QKV);
  attn_k<1><<<128*8*2, blk, 0, stream>>>(QKV, PB, AO, 256, 2, 256, 1);
  gemm_k<EPI_PROJ1><<<dim3(2, M/128), blk, 0, stream>>>(
      AO, T0r, 256, 256, nullptr, XN, nullptr, X, nullptr);

  // ln3 + column-wise gated MHSA (no bias), output transposed back for free
  ln_k<<<M/8, blk, 0, stream>>>(X, XN, g3, be3);
  gemm_k<EPI_QKV><<<dim3(8, M/128), blk, 0, stream>>>(
      XN, TQc, 1024, 256, nullptr, nullptr, nullptr, nullptr, QKV);
  attn_k<0><<<256*8*1, blk, 0, stream>>>(QKV, nullptr, AO, 128, 1, 1, 256);
  gemm_k<EPI_PROJ2><<<dim3(2, M/128), blk, 0, stream>>>(
      AO, T0c, 256, 256, nullptr, nullptr, nullptr, X, XN);

  // transition MLP: relu(x@W1+b1)@W2 + b2 + x
  gemm_k<EPI_MLP1><<<dim3(8, M/128), blk, 0, stream>>>(
      XN, T1, 1024, 256, b1, nullptr, nullptr, nullptr, QKV);
  gemm_k<EPI_MLP2><<<dim3(2, M/128), blk, 0, stream>>>(
      QKV, T2, 256, 1024, b2, nullptr, X, out, nullptr);
}
